// DotProductAttention_88708254531980
// MI455X (gfx1250) — compile-verified, matrix operand order checked by reading, not run on hardware
//
#include <hip/hip_runtime.h>

typedef __attribute__((ext_vector_type(16))) _Float16 v16h;
typedef __attribute__((ext_vector_type(8)))  _Float16 v8h;
typedef __attribute__((ext_vector_type(4)))  _Float16 v4h;
typedef __attribute__((ext_vector_type(8)))  float    v8f;

#define BATCH   64
#define SEQ     1024
#define DIM     64
#define KCHUNK  64
#define WAVES   8
#define QPB     (WAVES * 16)   // 128 queries per block
#define LDSPAD  72             // padded row stride in halves (144B: 16B-aligned, bank-staggered)
#define NEGF    (-1.0e6f)
// 1/sqrt(64) * log2(e): softmax runs in exp2 domain -> bare v_exp_f32
#define SCALE_LOG2E (0.125f * 1.44269504088896340736f)

// Raw v_exp_f32: args are always <= 0 here; underflow-to-zero is the desired
// behavior, so skip libm's 7-op denormal-range fixup.
#if __has_builtin(__builtin_amdgcn_exp2f)
#define FAST_EXP2(x) __builtin_amdgcn_exp2f(x)
#else
#define FAST_EXP2(x) exp2f(x)
#endif
#if __has_builtin(__builtin_amdgcn_rcpf)
#define FAST_RCP(x) __builtin_amdgcn_rcpf(x)
#else
#define FAST_RCP(x) (1.0f / (x))
#endif

__global__ __launch_bounds__(256)
void flash_attn_st_f16wmma_kernel(const float* __restrict__ Q,
                                  const float* __restrict__ K,
                                  const float* __restrict__ V,
                                  const int*   __restrict__ vlens,
                                  float*       __restrict__ O)
{
    // Double-buffered staging: one barrier per chunk, global latency overlapped
    __shared__ __align__(16) _Float16 Ksh[2][KCHUNK * LDSPAD];  // [buf][key][d]
    __shared__ __align__(16) _Float16 Vsh[2][DIM * LDSPAD];     // [buf][d][key]

    const int tid  = threadIdx.x;
    const int lane = tid & 31;
    const int wv   = tid >> 5;
    const int g    = lane >> 4;   // half-wave group
    const int ln   = lane & 15;   // this lane's query column (and A-tile row)

    const int b    = blockIdx.x >> 3;                  // SEQ/QPB == 8
    const int q0   = (blockIdx.x & 7) * QPB + wv * 16;
    const int vlen = vlens[b];

    // Per-thread staging coordinates (4 float4 groups per thread per chunk)
    const int skey0 = tid >> 4;          // + 16*i
    const int sdg   = (tid & 15) * 4;

    // ---- Q^T in WMMA B-layout (lane = query col, elements = d rows), scale folded ----
    v16h qb[2];
    {
        const float* qr = Q + ((size_t)b * SEQ + (q0 + ln)) * DIM;
        #pragma unroll
        for (int c = 0; c < 2; ++c)
            #pragma unroll
            for (int h = 0; h < 2; ++h) {
                const int dbase = 32 * c + 16 * h + 8 * g;
                #pragma unroll
                for (int j = 0; j < 8; ++j)
                    qb[c][h * 8 + j] = (_Float16)(qr[dbase + j] * SCALE_LOG2E);
            }
    }

    // All-ones A tile: row-sum of P^T via the matrix pipe
    v16h onesA;
    #pragma unroll
    for (int e = 0; e < 16; ++e) onesA[e] = (_Float16)1.0f;

    v8f oT[4];
    {
        v8f z = {};
        #pragma unroll
        for (int t = 0; t < 4; ++t) oT[t] = z;
    }
    float m = -3.0e38f, l = 0.0f;

    const int nchunks = (vlen + KCHUNK - 1) / KCHUNK;   // uniform per block

    // ---- Prologue: stage chunk 0 into buffer 0 ----
    {
        const size_t base = (size_t)b * SEQ * DIM;
        #pragma unroll
        for (int i = 0; i < 4; ++i) {
            const int key = skey0 + 16 * i;
            const float4 kq = *(const float4*)(K + base + (size_t)key * DIM + sdg);
            const float4 vq = *(const float4*)(V + base + (size_t)key * DIM + sdg);
            v4h hk;
            hk[0] = (_Float16)kq.x; hk[1] = (_Float16)kq.y;
            hk[2] = (_Float16)kq.z; hk[3] = (_Float16)kq.w;
            *(v4h*)&Ksh[0][key * LDSPAD + sdg] = hk;
            const float vv[4] = { vq.x, vq.y, vq.z, vq.w };
            #pragma unroll
            for (int j = 0; j < 4; ++j)
                Vsh[0][(sdg + j) * LDSPAD + key] = (_Float16)vv[j];
        }
    }
    __syncthreads();

    for (int ci = 0; ci < nchunks; ++ci) {
        const int  kc0      = ci * KCHUNK;
        const int  buf      = ci & 1;
        const bool has_next = (ci + 1 < nchunks);

        // ---- Issue next chunk's global loads NOW (land during compute) ----
        float4 nk[4], nv[4];
        if (has_next) {
            const size_t nbase = ((size_t)b * SEQ + kc0 + KCHUNK) * DIM;
            #pragma unroll
            for (int i = 0; i < 4; ++i) {
                const int key = skey0 + 16 * i;
                nk[i] = *(const float4*)(K + nbase + (size_t)key * DIM + sdg);
                nv[i] = *(const float4*)(V + nbase + (size_t)key * DIM + sdg);
            }
            if (ci + 2 < nchunks) {   // keep L2 warm two chunks ahead
                const size_t pbase = ((size_t)b * SEQ + kc0 + 2 * KCHUNK) * DIM + tid * 16;
                __builtin_prefetch(K + pbase, 0, 1);
                __builtin_prefetch(V + pbase, 0, 1);
            }
        }

        // ---- S^T = K * Q^T : 4 key tiles x 2 D-chunks ----
        v8f s[4];
        #pragma unroll
        for (int nt = 0; nt < 4; ++nt) {
            v8f acc = {};
            #pragma unroll
            for (int c = 0; c < 2; ++c) {
                const _Float16* kr = &Ksh[buf][(nt * 16 + ln) * LDSPAD + 32 * c + 8 * g];
                const v8h lo = *(const v8h*)kr;
                const v8h hi = *(const v8h*)(kr + 16);
                v16h ka;
                #pragma unroll
                for (int j = 0; j < 8; ++j) { ka[j] = lo[j]; ka[8 + j] = hi[j]; }
                acc = __builtin_amdgcn_wmma_f32_16x16x32_f16(
                          false, ka, false, qb[c], (short)0, acc, false, false);
            }
            s[nt] = acc;
        }

        // ---- Sequence mask: only the (single, partial) tail chunk pays ----
        if (kc0 + KCHUNK > vlen) {
            #pragma unroll
            for (int nt = 0; nt < 4; ++nt)
                #pragma unroll
                for (int r = 0; r < 8; ++r)
                    if (kc0 + nt * 16 + 8 * g + r >= vlen) s[nt][r] = NEGF;
        }

        // ---- Online softmax (log2 domain): one query per lane ----
        float tmax = s[0][0];
        #pragma unroll
        for (int nt = 0; nt < 4; ++nt)
            #pragma unroll
            for (int r = 0; r < 8; ++r) tmax = fmaxf(tmax, s[nt][r]);
        tmax = fmaxf(tmax, __shfl_xor(tmax, 16, 32));   // combine the two key halves
        const float mnew  = fmaxf(m, tmax);
        const float alpha = FAST_EXP2(m - mnew);
        m  = mnew;
        l *= alpha;
        #pragma unroll
        for (int t4 = 0; t4 < 4; ++t4)
            #pragma unroll
            for (int r = 0; r < 8; ++r) oT[t4][r] *= alpha;

        // ---- P^T B-tiles straight from registers (C/D layout == B layout) ----
        v16h pb[2];
        #pragma unroll
        for (int a = 0; a < 2; ++a)
            #pragma unroll
            for (int e = 0; e < 16; ++e) {
                const int nt = 2 * a + (e >> 3);
                const int r  = e & 7;
                pb[a][e] = (_Float16)FAST_EXP2(s[nt][r] - mnew);
            }

        // ---- Row-sum of P via ones-WMMA (matrix pipe) ----
        v8f lacc = {};
        #pragma unroll
        for (int a = 0; a < 2; ++a)
            lacc = __builtin_amdgcn_wmma_f32_16x16x32_f16(
                       false, onesA, false, pb[a], (short)0, lacc, false, false);

        // ---- O^T += V^T * P^T : 4 D tiles x 2 key chunks ----
        #pragma unroll
        for (int t4 = 0; t4 < 4; ++t4)
            #pragma unroll
            for (int a = 0; a < 2; ++a) {
                const _Float16* vr = &Vsh[buf][(t4 * 16 + ln) * LDSPAD + 32 * a + 8 * g];
                const v8h lo = *(const v8h*)vr;
                const v8h hi = *(const v8h*)(vr + 16);
                v16h va;
                #pragma unroll
                for (int j = 0; j < 8; ++j) { va[j] = lo[j]; va[8 + j] = hi[j]; }
                oT[t4] = __builtin_amdgcn_wmma_f32_16x16x32_f16(
                             false, va, false, pb[a], (short)0, oT[t4], false, false);
            }

        l += lacc[0];   // all 8 rows of the ones-product are identical

        // ---- Convert + store the prefetched next chunk into the other buffer ----
        if (has_next) {
            const int nb = buf ^ 1;
            #pragma unroll
            for (int i = 0; i < 4; ++i) {
                const int key = skey0 + 16 * i;
                v4h hk;
                hk[0] = (_Float16)nk[i].x; hk[1] = (_Float16)nk[i].y;
                hk[2] = (_Float16)nk[i].z; hk[3] = (_Float16)nk[i].w;
                *(v4h*)&Ksh[nb][key * LDSPAD + sdg] = hk;
                const float vv[4] = { nv[i].x, nv[i].y, nv[i].z, nv[i].w };
                #pragma unroll
                for (int j = 0; j < 4; ++j)
                    Vsh[nb][(sdg + j) * LDSPAD + key] = (_Float16)vv[j];
            }
            __syncthreads();   // uniform: protects both buffers across waves
        }
    }

    // ---- Epilogue: O = O^T / l, 32B-contiguous float4 stores per lane ----
    const float invl = FAST_RCP(l);
    float* orow = O + ((size_t)b * SEQ + (q0 + ln)) * DIM;
    #pragma unroll
    for (int t4 = 0; t4 < 4; ++t4) {
        float4 s0, s1;
        s0.x = oT[t4][0] * invl; s0.y = oT[t4][1] * invl;
        s0.z = oT[t4][2] * invl; s0.w = oT[t4][3] * invl;
        s1.x = oT[t4][4] * invl; s1.y = oT[t4][5] * invl;
        s1.z = oT[t4][6] * invl; s1.w = oT[t4][7] * invl;
        *(float4*)&orow[t4 * 16 + 8 * g]     = s0;
        *(float4*)&orow[t4 * 16 + 8 * g + 4] = s1;
    }
}

extern "C" void kernel_launch(void* const* d_in, const int* in_sizes, int n_in,
                              void* d_out, int out_size, void* d_ws, size_t ws_size,
                              hipStream_t stream) {
    const float* Q  = (const float*)d_in[0];
    const float* K  = (const float*)d_in[1];
    const float* V  = (const float*)d_in[2];
    const int*   vl = (const int*)d_in[3];
    float*       Op = (float*)d_out;

    dim3 grid(BATCH * (SEQ / QPB));   // 512 blocks
    dim3 block(256);                  // 8 waves of 32
    flash_attn_st_f16wmma_kernel<<<grid, block, 0, stream>>>(Q, K, V, vl, Op);
}
